// LabelSememeFusionPrompt_70738111365750
// MI455X (gfx1250) — matrix operand order repaired and checked by reading, not compiled, not run on hardware
//
#include <hip/hip_runtime.h>
#include <hip/hip_bf16.h>

// ---------------------------------------------------------------------------
// Types for CDNA5 WMMA (wave32)
// ---------------------------------------------------------------------------
typedef __attribute__((ext_vector_type(16))) __bf16 v16bf;
typedef __attribute__((ext_vector_type(8)))  float  v8f;

union ABv { uint4 u[2]; v16bf v; };

__device__ __forceinline__ unsigned short f2b(float x) {
    unsigned int u = __float_as_uint(x);
    u += 0x7FFFu + ((u >> 16) & 1u);   // round-to-nearest-even
    return (unsigned short)(u >> 16);
}

__device__ __forceinline__ int rmap(int r, int mode) {
    // mode 0: identity. mode 1: label half rows. mode 2: text half rows.
    if (mode == 1) return ((r >> 8) << 9) + (r & 255);
    if (mode == 2) return ((r >> 8) << 9) + 256 + (r & 255);
    return r;
}

__device__ __forceinline__ float sigmoidf_(float x) {
    return 1.0f / (1.0f + __expf(-x));
}

// ---------------------------------------------------------------------------
// Generic elementwise helpers
// ---------------------------------------------------------------------------
__global__ void cast_f32_bf16(const float* __restrict__ in,
                              unsigned short* __restrict__ out, int n) {
    for (int i = blockIdx.x * blockDim.x + threadIdx.x; i < n;
         i += gridDim.x * blockDim.x)
        out[i] = f2b(in[i]);
}

__global__ void bias_combine(const float* __restrict__ a,
                             const float* __restrict__ b,
                             float* __restrict__ out, int n) {
    int i = blockIdx.x * blockDim.x + threadIdx.x;
    if (i < n) out[i] = a[i] + b[i];
}

__global__ void copy_f32(const float* __restrict__ src,
                         float* __restrict__ dst, int n) {
    for (int i = blockIdx.x * blockDim.x + threadIdx.x; i < n;
         i += gridDim.x * blockDim.x)
        dst[i] = src[i];
}

// label_T[d*128 + l] = label[l*1024 + d]   (bf16 -> bf16)
__global__ void transpose_label(const unsigned short* __restrict__ in,
                                unsigned short* __restrict__ out) {
    int i = blockIdx.x * blockDim.x + threadIdx.x;   // i < 128*1024
    if (i >= 128 * 1024) return;
    int l = i >> 10, d = i & 1023;
    out[d * 128 + l] = in[i];
}

// wts = pt + text_feature[b] broadcast, output bf16 (4096 x 1024)
__global__ void build_wts(const float* __restrict__ prompt_fc,
                          const float* __restrict__ text_f,
                          unsigned short* __restrict__ out) {
    for (long long i = blockIdx.x * blockDim.x + threadIdx.x;
         i < 4096LL * 1024; i += (long long)gridDim.x * blockDim.x) {
        int r = (int)(i >> 10), d = (int)(i & 1023);
        int b = r >> 8;
        float v = prompt_fc[(size_t)rmap(r, 2) * 1024 + d] +
                  text_f[b * 1024 + d];
        out[i] = f2b(v);
    }
}

// Row softmax over 128 columns. One wave32 per row. out = bf16.
__global__ void softmax128(const float* __restrict__ in,
                           unsigned short* __restrict__ out) {
    int row  = blockIdx.x;
    int lane = threadIdx.x;            // 0..31
    const float* p = in + (size_t)row * 128;
    float v0 = p[lane], v1 = p[lane + 32], v2 = p[lane + 64], v3 = p[lane + 96];
    float m = fmaxf(fmaxf(v0, v1), fmaxf(v2, v3));
    for (int off = 16; off >= 1; off >>= 1) m = fmaxf(m, __shfl_xor(m, off, 32));
    float e0 = __expf(v0 - m), e1 = __expf(v1 - m),
          e2 = __expf(v2 - m), e3 = __expf(v3 - m);
    float s = e0 + e1 + e2 + e3;
    for (int off = 16; off >= 1; off >>= 1) s += __shfl_xor(s, off, 32);
    float r = 1.0f / s;
    unsigned short* q = out + (size_t)row * 128;
    q[lane]      = f2b(e0 * r);
    q[lane + 32] = f2b(e1 * r);
    q[lane + 64] = f2b(e2 * r);
    q[lane + 96] = f2b(e3 * r);
}

// ---------------------------------------------------------------------------
// WMMA bf16 GEMM:  out = act( A[M,K] @ B[N,K]^T + bias + resid )
// Block = 256 threads = 8 waves.  Block tile = 32 x 512.  Each wave computes a
// 32x64 tile (8 accumulators): two A frags (staged in LDS via async-to-LDS,
// double-buffered) reuse four B frags loaded from global (L2-resident weights).
// Requires: N % 64 == 0, K % 32 == 0; M % 16 == 0 (M=16 handled by clamping).
// act: 0 = none, 1 = tanh, 2 = relu
// ---------------------------------------------------------------------------
__global__ __launch_bounds__(256) void wmma_gemm_bf16(
    const unsigned short* __restrict__ A, int lda, int amap,
    const unsigned short* __restrict__ B, int ldb,
    const float* __restrict__ bias,
    const float* __restrict__ resid, int ldr, int rmode,
    float* __restrict__ outf, unsigned short* __restrict__ outb,
    int ldo, int omode, int M, int N, int K, int act)
{
    __shared__ __align__(16) unsigned short astage[2][32 * 32];  // 2 x 2KB

    const int tid  = threadIdx.x;
    const int lane = tid & 31;
    const int wv   = tid >> 5;
    const int m0   = blockIdx.x * 32;
    const int n0   = (blockIdx.y * 8 + wv) * 64;
    const bool active = (n0 < N);          // wave-uniform
    const int hl  = lane >> 4;
    const int l15 = lane & 15;
    const int nk  = K >> 5;

    // ---- async A staging: 256 threads x 8B cover the 32x32 bf16 chunk ----
    const int srow = tid >> 3;             // 0..31
    const int scol = (tid & 7) * 4;        // element offset within k-chunk
    int grow = m0 + srow; if (grow > M - 1) grow = M - 1;
    const unsigned short* gsrc = A + (size_t)rmap(grow, amap) * lda + scol;
    // Low 32 bits of an LDS-aperture flat address are the LDS byte offset.
    unsigned lds_off[2];
    lds_off[0] = (unsigned)(uintptr_t)&astage[0][srow * 32 + scol];
    lds_off[1] = (unsigned)(uintptr_t)&astage[1][srow * 32 + scol];

#define STAGE_A(kc, buf)                                                    \
    asm volatile("global_load_async_to_lds_b64 %0, %1, off"                 \
                 :: "v"(lds_off[buf]), "v"(gsrc + (kc) * 32) : "memory")

    const unsigned short* bp[4];
#pragma unroll
    for (int tt = 0; tt < 4; ++tt)
        bp[tt] = B + (size_t)(n0 + tt * 16 + l15) * ldb + hl * 8;

    v8f acc[2][4];
#pragma unroll
    for (int mt = 0; mt < 2; ++mt)
#pragma unroll
        for (int tt = 0; tt < 4; ++tt)
#pragma unroll
            for (int i = 0; i < 8; ++i) acc[mt][tt][i] = 0.f;

    STAGE_A(0, 0);
    for (int kc = 0; kc < nk; ++kc) {
        const int buf = kc & 1;
        if (kc + 1 < nk) {
            STAGE_A(kc + 1, buf ^ 1);
            asm volatile("s_wait_asynccnt 0x1" ::: "memory");
        } else {
            asm volatile("s_wait_asynccnt 0x0" ::: "memory");
        }
        __syncthreads();                   // staged A chunk visible to block

        if (active) {
            const int k0 = kc * 32;
            // A fragments from LDS (ds_load_b128 x2 per M-tile)
            ABv a0, a1;
            const unsigned short* abase = &astage[buf][0];
            a0.u[0] = *(const uint4*)(abase + (l15)      * 32 + hl * 8);
            a0.u[1] = *(const uint4*)(abase + (l15)      * 32 + 16 + hl * 8);
            a1.u[0] = *(const uint4*)(abase + (16 + l15) * 32 + hl * 8);
            a1.u[1] = *(const uint4*)(abase + (16 + l15) * 32 + 16 + hl * 8);

            ABv b0, b1, b2, b3;
            b0.u[0] = *(const uint4*)(bp[0] + k0); b0.u[1] = *(const uint4*)(bp[0] + k0 + 16);
            b1.u[0] = *(const uint4*)(bp[1] + k0); b1.u[1] = *(const uint4*)(bp[1] + k0 + 16);
            b2.u[0] = *(const uint4*)(bp[2] + k0); b2.u[1] = *(const uint4*)(bp[2] + k0 + 16);
            b3.u[0] = *(const uint4*)(bp[3] + k0); b3.u[1] = *(const uint4*)(bp[3] + k0 + 16);
            __builtin_prefetch((const void*)(bp[0] + k0 + 64), 0, 0);
            __builtin_prefetch((const void*)(bp[2] + k0 + 64), 0, 0);

            acc[0][0] = __builtin_amdgcn_wmma_f32_16x16x32_bf16(false, a0.v, false, b0.v, (short)0, acc[0][0], false, false);
            acc[0][1] = __builtin_amdgcn_wmma_f32_16x16x32_bf16(false, a0.v, false, b1.v, (short)0, acc[0][1], false, false);
            acc[0][2] = __builtin_amdgcn_wmma_f32_16x16x32_bf16(false, a0.v, false, b2.v, (short)0, acc[0][2], false, false);
            acc[0][3] = __builtin_amdgcn_wmma_f32_16x16x32_bf16(false, a0.v, false, b3.v, (short)0, acc[0][3], false, false);
            acc[1][0] = __builtin_amdgcn_wmma_f32_16x16x32_bf16(false, a1.v, false, b0.v, (short)0, acc[1][0], false, false);
            acc[1][1] = __builtin_amdgcn_wmma_f32_16x16x32_bf16(false, a1.v, false, b1.v, (short)0, acc[1][1], false, false);
            acc[1][2] = __builtin_amdgcn_wmma_f32_16x16x32_bf16(false, a1.v, false, b2.v, (short)0, acc[1][2], false, false);
            acc[1][3] = __builtin_amdgcn_wmma_f32_16x16x32_bf16(false, a1.v, false, b3.v, (short)0, acc[1][3], false, false);
        }
        __syncthreads();                   // reads done before buf is reused
    }
#undef STAGE_A

    if (!active) return;

#pragma unroll
    for (int mt = 0; mt < 2; ++mt) {
        if (m0 + mt * 16 >= M) break;      // M=16 edge
#pragma unroll
        for (int tt = 0; tt < 4; ++tt) {
            int col = n0 + tt * 16 + l15;
#pragma unroll
            for (int v = 0; v < 8; ++v) {
                int row = m0 + mt * 16 + v + 8 * hl;   // logical row
                float x = acc[mt][tt][v];
                if (bias)  x += bias[col];
                if (resid) x += resid[(size_t)rmap(row, rmode) * ldr + col];
                if (act == 1) x = tanhf(x);
                else if (act == 2) x = fmaxf(x, 0.f);
                size_t oidx = (size_t)rmap(row, omode) * ldo + col;
                if (outf) outf[oidx] = x;
                if (outb) outb[oidx] = f2b(x);
            }
        }
    }
}

// ---------------------------------------------------------------------------
// Persistent bidirectional-LSTM recurrence kernel (one layer).
// grid.x = 2 (dir 0 = forward, dir 1 = backward), block = 1024 = 32 waves.
// gates_x: precomputed x@Wih^T + (bih+bhh), layout [b*512+t][2048] f32.
// whh: bf16 [2048][512].  Wave w owns hidden cols [16w,16w+16); its four gate
// tiles (i,f,g,o) are n = w, w+32, w+64, w+96, so i/f/g/o for each (batch,
// hidden) land in the same lane/slot — cell state c stays in registers.
// h is exchanged between steps via a 16KB bf16 LDS tile read as WMMA A ops.
// Output: xout[b*512+t][dir*512 + hidden] bf16  (B=16, S=512, feat=1024)
// ---------------------------------------------------------------------------
__global__ __launch_bounds__(1024) void lstm_layer_kernel(
    const float* __restrict__ gates_fwd, const float* __restrict__ gates_bwd,
    const unsigned short* __restrict__ whh_fwd,
    const unsigned short* __restrict__ whh_bwd,
    unsigned short* __restrict__ xout)
{
    const int dir = blockIdx.x;
    const float* gates = dir ? gates_bwd : gates_fwd;
    const unsigned short* whh = dir ? whh_bwd : whh_fwd;

    __shared__ __align__(16) unsigned short h_lds[16 * 512];   // 16KB bf16

    const int tid  = threadIdx.x;
    const int lane = tid & 31;
    const int wv   = tid >> 5;          // 0..31 -> hidden cols 16*wv..
    const int hl   = lane >> 4;
    const int l15  = lane & 15;

    for (int i = tid; i < 16 * 512; i += 1024) h_lds[i] = 0;
    __syncthreads();

    float c[8];
#pragma unroll
    for (int v = 0; v < 8; ++v) c[v] = 0.f;

    const unsigned short* hrow = &h_lds[l15 * 512];
    const unsigned short* bp[4];
#pragma unroll
    for (int g = 0; g < 4; ++g)
        bp[g] = whh + (size_t)((wv + g * 32) * 16 + l15) * 512 + hl * 8;

    for (int s = 0; s < 512; ++s) {
        const int t = dir ? (511 - s) : s;

        v8f acc[4];
#pragma unroll
        for (int g = 0; g < 4; ++g)
#pragma unroll
            for (int i = 0; i < 8; ++i) acc[g][i] = 0.f;

        // h(t-1) @ Whh^T for this wave's four gate tiles: 64 WMMAs / wave / step
        for (int kc = 0; kc < 16; ++kc) {
            const int k0 = kc * 32;
            ABv a;
            a.u[0] = *(const uint4*)(hrow + k0 + hl * 8);        // ds_load_b128
            a.u[1] = *(const uint4*)(hrow + k0 + 16 + hl * 8);
#pragma unroll
            for (int g = 0; g < 4; ++g) {
                ABv b;
                b.u[0] = *(const uint4*)(bp[g] + k0);
                b.u[1] = *(const uint4*)(bp[g] + k0 + 16);
                __builtin_prefetch((const void*)(bp[g] + k0 + 64), 0, 0);
                acc[g] = __builtin_amdgcn_wmma_f32_16x16x32_bf16(
                    false, a.v, false, b.v, (short)0, acc[g], false, false);
            }
        }

        // Gate nonlinearity + cell update (per-lane: batch = v+8*hl,
        // hidden = 16*wv + l15).  Add precomputed x-gates (bias folded in).
        float hval[8];
        const int hcol = 16 * wv + l15;
#pragma unroll
        for (int v = 0; v < 8; ++v) {
            const int b = v + 8 * hl;
            const float* gx = gates + ((size_t)b * 512 + t) * 2048;
            float gi = acc[0][v] + gx[hcol];
            float gf = acc[1][v] + gx[hcol + 512];
            float gg = acc[2][v] + gx[hcol + 1024];
            float go = acc[3][v] + gx[hcol + 1536];
            float cn = sigmoidf_(gf) * c[v] + sigmoidf_(gi) * tanhf(gg);
            c[v] = cn;
            hval[v] = sigmoidf_(go) * tanhf(cn);
        }

        __syncthreads();   // everyone done reading h(t-1) from LDS
#pragma unroll
        for (int v = 0; v < 8; ++v) {
            const int b = v + 8 * hl;
            unsigned short hb = f2b(hval[v]);
            h_lds[b * 512 + hcol] = hb;                           // ds_store
            xout[((size_t)b * 512 + t) * 1024 + dir * 512 + hcol] = hb;
        }
        __syncthreads();   // h(t) visible for next step's WMMA reads
    }
}

// ---------------------------------------------------------------------------
// Host-side orchestration
// ---------------------------------------------------------------------------
extern "C" void kernel_launch(void* const* d_in, const int* in_sizes, int n_in,
                              void* d_out, int out_size, void* d_ws, size_t ws_size,
                              hipStream_t stream) {
    (void)in_sizes; (void)n_in; (void)out_size; (void)ws_size;
    constexpr int B = 16, S = 512, D = 1024, L = 128, HH = 512, G4 = 2048;
    constexpr int BS = B * S;            // 8192
    constexpr int HM = B * (S / 2);      // 4096

    const float* prompt  = (const float*)d_in[0];
    const float* lbl     = (const float*)d_in[1];
    const float* txt     = (const float*)d_in[2];
    const float* fc1w    = (const float*)d_in[3];
    const float* fc2w    = (const float*)d_in[4];
    const float* fc3w    = (const float*)d_in[5];
    const float* fc4w    = (const float*)d_in[6];
    const float* fc5w    = (const float*)d_in[7];
    const float* fc5b    = (const float*)d_in[8];
    const float* mlp1w   = (const float*)d_in[9];
    const float* mlp1b   = (const float*)d_in[10];
    const float* mlp2w   = (const float*)d_in[11];
    const float* mlp2b   = (const float*)d_in[12];
    // layer0: f = 13..16, b = 17..20 ; layer1: f = 21..24, b = 25..28
    const float* wih[4]  = { (const float*)d_in[13], (const float*)d_in[17],
                             (const float*)d_in[21], (const float*)d_in[25] };
    const float* whh[4]  = { (const float*)d_in[14], (const float*)d_in[18],
                             (const float*)d_in[22], (const float*)d_in[26] };
    const float* bih[4]  = { (const float*)d_in[15], (const float*)d_in[19],
                             (const float*)d_in[23], (const float*)d_in[27] };
    const float* bhh[4]  = { (const float*)d_in[16], (const float*)d_in[20],
                             (const float*)d_in[24], (const float*)d_in[28] };
    float* outp = (float*)d_out;

    // -------- deterministic workspace carve-out --------
    char* ws = (char*)d_ws;
    size_t off = 0;
    auto alloc = [&](size_t bytes) -> char* {
        char* p = ws + off;
        off = (off + bytes + 255) & ~(size_t)255;
        return p;
    };
    typedef unsigned short us;
    us* fc1B  = (us*)alloc((size_t)D * D * 2);
    us* fc2B  = (us*)alloc((size_t)D * D * 2);
    us* fc3B  = (us*)alloc((size_t)D * D * 2);
    us* fc4B  = (us*)alloc((size_t)D * D * 2);
    us* fc5B  = (us*)alloc((size_t)D * D * 2);
    us* m1B   = (us*)alloc((size_t)D * D * 2);
    us* m2B   = (us*)alloc((size_t)D * D * 2);
    us* wihB[4]; us* whhB[4]; float* cb[4];
    for (int i = 0; i < 4; ++i) {
        wihB[i] = (us*)alloc((size_t)G4 * D * 2);
        whhB[i] = (us*)alloc((size_t)G4 * HH * 2);
        cb[i]   = (float*)alloc((size_t)G4 * 4);
    }
    us*    promptB = (us*)alloc((size_t)BS * D * 2);
    us*    lblInB  = (us*)alloc((size_t)L * D * 2);
    us*    txtInB  = (us*)alloc((size_t)B * D * 2);
    float* pfcF    = (float*)alloc((size_t)BS * D * 4);
    us*    pfcB    = (us*)alloc((size_t)BS * D * 2);
    us*    lblFB   = (us*)alloc((size_t)L * D * 2);     // label_feature bf16 (L x D)
    us*    lblTB   = (us*)alloc((size_t)D * L * 2);     // transposed (D x L)
    float* txtF    = (float*)alloc((size_t)B * D * 4);
    float* scoresF = (float*)alloc((size_t)HM * L * 4);
    us*    attnB   = (us*)alloc((size_t)HM * L * 2);
    us*    wlsB    = (us*)alloc((size_t)HM * D * 2);
    us*    wtsB    = (us*)alloc((size_t)HM * D * 2);
    us*    fusedB  = (us*)alloc((size_t)BS * D * 2);
    float* gates0  = (float*)alloc((size_t)BS * G4 * 4);
    float* gates1  = (float*)alloc((size_t)BS * G4 * 4);
    us*    l0outB  = (us*)alloc((size_t)BS * D * 2);
    us*    l1outB  = (us*)alloc((size_t)BS * D * 2);
    us*    hidB    = (us*)alloc((size_t)BS * D * 2);

    auto cast = [&](const float* src, us* dst, long long n) {
        int blk = 256;
        int grd = (int)((n + blk - 1) / blk); if (grd > 65535) grd = 65535;
        cast_f32_bf16<<<grd, blk, 0, stream>>>(src, dst, (int)n);
    };
    auto gemm = [&](const us* A, int lda, int amap, const us* Bm, int ldb,
                    const float* bias, const float* resid, int ldr, int rmode,
                    float* outf, us* outb, int ldo, int omode,
                    int M, int N, int K, int act) {
        dim3 g((M + 31) / 32, (N + 511) / 512);
        wmma_gemm_bf16<<<g, 256, 0, stream>>>(A, lda, amap, Bm, ldb, bias,
                                              resid, ldr, rmode, outf, outb,
                                              ldo, omode, M, N, K, act);
    };

    // -------- 1. casts + combined biases --------
    cast(fc1w, fc1B, (long long)D * D);  cast(fc2w, fc2B, (long long)D * D);
    cast(fc3w, fc3B, (long long)D * D);  cast(fc4w, fc4B, (long long)D * D);
    cast(fc5w, fc5B, (long long)D * D);  cast(mlp1w, m1B, (long long)D * D);
    cast(mlp2w, m2B, (long long)D * D);
    for (int i = 0; i < 4; ++i) {
        cast(wih[i], wihB[i], (long long)G4 * D);
        cast(whh[i], whhB[i], (long long)G4 * HH);
        bias_combine<<<(G4 + 255) / 256, 256, 0, stream>>>(bih[i], bhh[i], cb[i], G4);
    }
    cast(prompt, promptB, (long long)BS * D);
    cast(lbl, lblInB, (long long)L * D);
    cast(txt, txtInB, (long long)B * D);

    // -------- 2. feature fusion --------
    // prompt_fc = prompt @ fc1^T
    gemm(promptB, D, 0, fc1B, D, nullptr, nullptr, 0, 0, pfcF, pfcB, D, 0, BS, D, D, 0);
    // label_feature = init_label @ fc2^T
    gemm(lblInB, D, 0, fc2B, D, nullptr, nullptr, 0, 0, nullptr, lblFB, D, 0, L, D, D, 0);
    transpose_label<<<(L * D + 255) / 256, 256, 0, stream>>>(lblFB, lblTB);
    // scores = pl @ label_feature^T
    gemm(pfcB, D, 1, lblFB, D, nullptr, nullptr, 0, 0, scoresF, nullptr, L, 0, HM, L, D, 0);
    softmax128<<<HM, 32, 0, stream>>>(scoresF, attnB);
    // wls = attn @ label_feature + pl
    gemm(attnB, L, 0, lblTB, L, nullptr, pfcF, D, 1, nullptr, wlsB, D, 0, HM, D, L, 0);
    // fused_label = tanh(wls @ fc5^T + fc5_b) -> fused[:, :256]
    gemm(wlsB, D, 0, fc5B, D, fc5b, nullptr, 0, 0, nullptr, fusedB, D, 1, HM, D, D, 1);
    // text_feature = init_text @ fc3^T
    gemm(txtInB, D, 0, fc3B, D, nullptr, nullptr, 0, 0, txtF, nullptr, D, 0, B, D, D, 0);
    build_wts<<<4096, 256, 0, stream>>>(pfcF, txtF, wtsB);
    // fused_text = tanh(wts @ fc4^T) -> fused[:, 256:]
    gemm(wtsB, D, 0, fc4B, D, nullptr, nullptr, 0, 0, nullptr, fusedB, D, 2, HM, D, D, 1);

    // -------- 3. LSTM layer 0 --------
    gemm(fusedB, D, 0, wihB[0], D, cb[0], nullptr, 0, 0, gates0, nullptr, G4, 0, BS, G4, D, 0);
    gemm(fusedB, D, 0, wihB[1], D, cb[1], nullptr, 0, 0, gates1, nullptr, G4, 0, BS, G4, D, 0);
    lstm_layer_kernel<<<2, 1024, 0, stream>>>(gates0, gates1, whhB[0], whhB[1], l0outB);

    // -------- 4. LSTM layer 1 --------
    gemm(l0outB, D, 0, wihB[2], D, cb[2], nullptr, 0, 0, gates0, nullptr, G4, 0, BS, G4, D, 0);
    gemm(l0outB, D, 0, wihB[3], D, cb[3], nullptr, 0, 0, gates1, nullptr, G4, 0, BS, G4, D, 0);
    lstm_layer_kernel<<<2, 1024, 0, stream>>>(gates0, gates1, whhB[2], whhB[3], l1outB);

    // -------- 5. MLP head --------
    gemm(l1outB, D, 0, m1B, D, mlp1b, nullptr, 0, 0, nullptr, hidB, D, 0, BS, D, D, 2);
    gemm(hidB, D, 0, m2B, D, mlp2b, nullptr, 0, 0, outp, nullptr, D, 0, BS, D, D, 0);

    // -------- 6. tuple tail: (out, init_text_feature, init_label_feature) ----
    copy_f32<<<(B * D + 255) / 256, 256, 0, stream>>>(txt, outp + (size_t)BS * D, B * D);
    copy_f32<<<(L * D + 255) / 256, 256, 0, stream>>>(lbl, outp + (size_t)BS * D + B * D, L * D);
}
